// WindowAttention_27822798143646
// MI455X (gfx1250) — matrix operand order repaired and checked by reading, not compiled, not run on hardware
//
#include <hip/hip_runtime.h>

// ---------------------------------------------------------------------------
// Window attention (Swin), fully fused per-window kernel for gfx1250 (MI455X).
// One workgroup (256 threads = 8 wave32) per window; f16 WMMA w/ f32 accum.
// ---------------------------------------------------------------------------

typedef __attribute__((ext_vector_type(16))) _Float16 v16h;
typedef __attribute__((ext_vector_type(8)))  float    v8f;

#define DIMC   192
#define NTOK   64
#define NHEAD  6
#define HDIM   32
#define QKVD   576          // 3*DIMC
#define SCALEQ 0.17677669529663687f   // 32^-0.5

// ---- WMMA fragment loaders (CDNA5 16x16x32 f16 layouts, wave32) -----------
// A (16x32 f16), row-major source with given stride:
//   lane L: M = L%16 ; halves i=0..15 -> K = 16*(L/16) + i (contiguous)
__device__ __forceinline__ v16h load_A(const _Float16* src, int m0, int k0,
                                       int stride, int lane) {
    const _Float16* p = src + (m0 + (lane & 15)) * stride + k0 + 16 * (lane >> 4);
    v16h a;
#pragma unroll
    for (int i = 0; i < 16; ++i) a[i] = p[i];
    return a;
}

// B (32x16 f16) from "B-transposed" storage (rows = N dim, cols = K dim):
//   lane L: N = L%16 ; halves j=0..15 -> K = 16*(L/16) + j (contiguous)
__device__ __forceinline__ v16h load_Bt(const _Float16* src, int n0, int k0,
                                        int stride, int lane) {
    const _Float16* p = src + (n0 + (lane & 15)) * stride + k0 + 16 * (lane >> 4);
    v16h b;
#pragma unroll
    for (int i = 0; i < 16; ++i) b[i] = p[i];
    return b;
}

__device__ __forceinline__ v8f wmma_f16(v16h a, v16h b, v8f c) {
    return __builtin_amdgcn_wmma_f32_16x16x32_f16(false, a, false, b,
                                                  (short)0, c, false, false);
}

__device__ __forceinline__ unsigned pack_h2(float a, float b) {
    union { _Float16 h[2]; unsigned u; } p;
    p.h[0] = (_Float16)a;
    p.h[1] = (_Float16)b;
    return p.u;
}

// ---- pre-pass: fp32 -> f16 weight conversion into workspace ---------------
__global__ void wa_convert_weights(const float* __restrict__ wq,
                                   const float* __restrict__ wp,
                                   _Float16* __restrict__ o) {
    const int nq = QKVD * DIMC;          // 110592
    const int total = nq + DIMC * DIMC;  // +36864
    for (int i = blockIdx.x * blockDim.x + threadIdx.x; i < total;
         i += gridDim.x * blockDim.x) {
        float v = (i < nq) ? wq[i] : wp[i - nq];
        o[i] = (_Float16)v;
    }
}

// ---- main fused kernel ----------------------------------------------------
__global__ __launch_bounds__(256)
void wa_fused_kernel(const float* __restrict__ x,
                     const float* __restrict__ b_qkv,
                     const float* __restrict__ b_proj,
                     const float* __restrict__ bias_table,
                     const _Float16* __restrict__ wq16,   // [576][192] f16
                     const _Float16* __restrict__ wp16,   // [192][192] f16
                     float* __restrict__ out) {
    extern __shared__ _Float16 smem[];
    _Float16* qkv   = smem;                 // 64*576 f16  (72 KB)
    _Float16* xtile = smem + 36864;         // 64*192 f16  (24 KB) -> reused as ctx
    _Float16* vt    = smem + 49152;         // [6][32][64] f16 (24 KB)
    _Float16* probs = smem + 61440;         // 64*64  f16  ( 8 KB)
    float*    scores = (float*)(smem + 65536); // 64*64 f32 (16 KB)
    _Float16* ctx = xtile;

    const int tid  = threadIdx.x;
    const int lane = tid & 31;
    const int wid  = tid >> 5;           // 8 waves
    const int b    = blockIdx.x;
    const float* xg = x + (size_t)b * (NTOK * DIMC);

    // ---- Stage 0: x -> f16 LDS tile (packed dword stores) ----------------
    {
        unsigned* xt32 = (unsigned*)xtile;
        for (int i = tid; i < NTOK * DIMC / 4; i += 256) {
            float4 f = ((const float4*)xg)[i];
            xt32[i * 2 + 0] = pack_h2(f.x, f.y);
            xt32[i * 2 + 1] = pack_h2(f.z, f.w);
        }
    }
    __syncthreads();

    // ---- Stage 1: qkv = x @ w_qkv^T + b ; scale q ; build vt -------------
    // 4 x 36 output tiles, 6 K-steps; 18 tiles per wave. nt is wave-uniform,
    // and the q/k/v column boundaries (192, 384) are tile-aligned, so the
    // q-scale and v-transpose predicates are scalar branches, not exec masks.
#pragma unroll 1
    for (int i = 0; i < 18; ++i) {
        int t = i * 8 + wid;
        int mt = t / 36, nt = t % 36;         // wave-uniform
        v8f acc = {};
#pragma unroll
        for (int kt = 0; kt < 6; ++kt) {
            v16h a  = load_A(xtile, mt * 16, kt * 32, DIMC, lane);
            v16h bf = load_Bt(wq16, nt * 16, kt * 32, DIMC, lane);
            acc = wmma_f16(a, bf, acc);
        }
        const int col   = nt * 16 + (lane & 15);
        const int rbase = mt * 16 + 8 * (lane >> 4);
        const float bias = b_qkv[col];
        const float qscale = (nt < 12) ? SCALEQ : 1.0f;   // uniform select
#pragma unroll
        for (int v = 0; v < 8; ++v) {
            _Float16 hv = (_Float16)((acc[v] + bias) * qscale);
            qkv[(rbase + v) * QKVD + col] = hv;
            acc[v] = (float)hv;               // keep rounded value for vt copy
        }
        if (nt >= 24) {                       // v part -> transposed copy
            const int hh = (col - 2 * DIMC) >> 5;
            const int e  = (col - 2 * DIMC) & 31;
            _Float16* vtc = vt + hh * (HDIM * NTOK) + e * NTOK + rbase;
#pragma unroll
            for (int v = 0; v < 8; ++v) vtc[v] = (_Float16)acc[v];
        }
    }
    __syncthreads();

    // ---- Stage 2: per-head attention -------------------------------------
#pragma unroll 1
    for (int h = 0; h < NHEAD; ++h) {
        // scores = q @ k^T : 4x4 tiles, K=32 (one step); 2 tiles per wave
#pragma unroll
        for (int i = 0; i < 2; ++i) {
            int t = wid * 2 + i;
            int mt = t >> 2, nt = t & 3;
            v16h a  = load_A(qkv + h * HDIM,         mt * 16, 0, QKVD, lane); // q
            v16h bf = load_Bt(qkv + DIMC + h * HDIM, nt * 16, 0, QKVD, lane); // k
            v8f acc = {};
            acc = wmma_f16(a, bf, acc);
            int col   = nt * 16 + (lane & 15);
            int rbase = mt * 16 + 8 * (lane >> 4);
#pragma unroll
            for (int v = 0; v < 8; ++v)
                scores[(rbase + v) * NTOK + col] = acc[v];
        }
        __syncthreads();

        // softmax over rows + relative-position bias (threads 0..63)
        if (tid < NTOK) {
            const int r = tid, yi = r >> 3, xi = r & 7;
            float vals[NTOK];
            float mx = -1e30f;
#pragma unroll
            for (int j = 0; j < NTOK; ++j) {
                int yj = j >> 3, xj = j & 7;
                int idx = (yi - yj + 7) * 15 + (xi - xj + 7);
                float s = scores[r * NTOK + j] + bias_table[idx * NHEAD + h];
                vals[j] = s;
                mx = fmaxf(mx, s);
            }
            float sum = 0.f;
#pragma unroll
            for (int j = 0; j < NTOK; ++j) {
                float e = __expf(vals[j] - mx);
                vals[j] = e;
                sum += e;
            }
            float inv = 1.0f / sum;
            unsigned* pr32 = (unsigned*)(probs + r * NTOK);
#pragma unroll
            for (int j = 0; j < NTOK / 2; ++j)
                pr32[j] = pack_h2(vals[2 * j] * inv, vals[2 * j + 1] * inv);
        }
        __syncthreads();

        // ctx_h = probs @ v_h : 4x2 tiles, 2 K-steps; 1 tile per wave
        {
            int mt = wid & 3, nt = wid >> 2;   // nt in 0..1
            v8f acc = {};
#pragma unroll
            for (int kt = 0; kt < 2; ++kt) {
                v16h a  = load_A(probs, mt * 16, kt * 32, NTOK, lane);
                v16h bf = load_Bt(vt + h * (HDIM * NTOK), nt * 16, kt * 32,
                                  NTOK, lane);
                acc = wmma_f16(a, bf, acc);
            }
            int e     = nt * 16 + (lane & 15);
            int rbase = mt * 16 + 8 * (lane >> 4);
#pragma unroll
            for (int v = 0; v < 8; ++v)
                ctx[(rbase + v) * DIMC + h * HDIM + e] = (_Float16)acc[v];
        }
        __syncthreads();
    }

    // ---- Stage 3: out = ctx @ w_proj^T + b_proj --------------------------
    // 4 x 12 tiles, 6 K-steps; 6 tiles per wave.
    float* og = out + (size_t)b * (NTOK * DIMC);
#pragma unroll 1
    for (int i = 0; i < 6; ++i) {
        int t = i * 8 + wid;
        int mt = t / 12, nt = t % 12;
        v8f acc = {};
#pragma unroll
        for (int kt = 0; kt < 6; ++kt) {
            v16h a  = load_A(ctx, mt * 16, kt * 32, DIMC, lane);
            v16h bf = load_Bt(wp16, nt * 16, kt * 32, DIMC, lane);
            acc = wmma_f16(a, bf, acc);
        }
        int col   = nt * 16 + (lane & 15);
        int rbase = mt * 16 + 8 * (lane >> 4);
        float bb = b_proj[col];
#pragma unroll
        for (int v = 0; v < 8; ++v)
            og[(rbase + v) * DIMC + col] = acc[v] + bb;
    }
}

// ---------------------------------------------------------------------------
extern "C" void kernel_launch(void* const* d_in, const int* in_sizes, int n_in,
                              void* d_out, int out_size, void* d_ws, size_t ws_size,
                              hipStream_t stream) {
    const float* x          = (const float*)d_in[0];
    const float* w_qkv      = (const float*)d_in[1];
    const float* b_qkv      = (const float*)d_in[2];
    const float* w_proj     = (const float*)d_in[3];
    const float* b_proj     = (const float*)d_in[4];
    const float* bias_table = (const float*)d_in[5];

    _Float16* w16 = (_Float16*)d_ws;                 // 110592 + 36864 halves
    wa_convert_weights<<<576, 256, 0, stream>>>(w_qkv, w_proj, w16);

    const int nblocks = in_sizes[0] / (NTOK * DIMC); // 4096
    const size_t lds_bytes = 65536 * sizeof(_Float16) + 64 * 64 * sizeof(float); // 147456
    wa_fused_kernel<<<nblocks, 256, lds_bytes, stream>>>(
        x, b_qkv, b_proj, bias_table, w16, w16 + QKVD * DIMC, (float*)d_out);
}
